// MHSA_83391085019732
// MI455X (gfx1250) — matrix operand order repaired and checked by reading, not compiled, not run on hardware
//
#include <hip/hip_runtime.h>
#include <hip/hip_bf16.h>

#define SEQ   2048
#define EMB   1024
#define HEADS 16
#define HDIM  64
#define BATCH 2
#define NROW  (BATCH * SEQ)   // 4096

typedef __bf16 bf16_t;
typedef __bf16        v2bf  __attribute__((ext_vector_type(2)));
typedef __bf16        v16bf __attribute__((ext_vector_type(16)));
typedef float         v8f   __attribute__((ext_vector_type(8)));
typedef unsigned int  v4u   __attribute__((ext_vector_type(4)));
typedef unsigned int  v8u   __attribute__((ext_vector_type(8)));

union FragU { v16bf v; v4u u4[2]; v8u u8; };

// ---- fp32 -> bf16 conversion helpers -------------------------------------
// Prefer the packed HW converter (1 VALU op / 2 values) when the toolchain
// declares it for gfx1250; otherwise fall back to the manual RNE sequence.
#if __has_builtin(__builtin_amdgcn_cvt_pk_bf16_f32)
__device__ __forceinline__ unsigned pk2bf(float a, float b) {
  v2bf p = __builtin_amdgcn_cvt_pk_bf16_f32(a, b);
  return __builtin_bit_cast(unsigned, p);
}
__device__ __forceinline__ bf16_t f2bf(float f) {
  v2bf p = __builtin_amdgcn_cvt_pk_bf16_f32(f, f);
  return p[0];
}
#else
__device__ __forceinline__ unsigned short f2bf_u(float f) {
  unsigned u = __builtin_bit_cast(unsigned, f);
  u += 0x7FFFu + ((u >> 16) & 1u);           // round-to-nearest-even
  return (unsigned short)(u >> 16);
}
__device__ __forceinline__ unsigned pk2bf(float a, float b) {
  return (unsigned)f2bf_u(a) | ((unsigned)f2bf_u(b) << 16);
}
__device__ __forceinline__ bf16_t f2bf(float f) {
  unsigned short h = f2bf_u(f);
  return __builtin_bit_cast(bf16_t, h);
}
#endif

// A fragment: 16x32 bf16, row-major source. lane m = l%16; hi = l/16.
// elements 0..7 -> K = k0 + 8*hi + e ; elements 8..15 -> K = k0 + 16 + 8*hi + (e-8)
__device__ __forceinline__ v16bf load_a_frag(const bf16_t* p0, int ld, int lane) {
  int m = lane & 15, hi = lane >> 4;
  const bf16_t* p = p0 + (size_t)m * ld + 8 * hi;
  FragU f;
  f.u4[0] = *(const v4u*)p;
  f.u4[1] = *(const v4u*)(p + 16);
  return f.v;
}

// B fragment: 32x16 (KxN), B[k][n] = src[n][k] (src row-major over n, contiguous k).
// lane n = l%16; hi = l/16; elements 0..7 -> K = k0 + 8*hi + e ; 8..15 -> K = k0 + 16 + 8*hi + (e-8)
__device__ __forceinline__ v16bf load_b_frag(const bf16_t* p0, int ld, int lane) {
  int n = lane & 15, hi = lane >> 4;
  const bf16_t* p = p0 + (size_t)n * ld + 8 * hi;
  FragU f;
  f.u4[0] = *(const v4u*)p;
  f.u4[1] = *(const v4u*)(p + 16);
  return f.v;
}

__device__ __forceinline__ v8f wmma_bf16(v16bf a, v16bf b, v8f c) {
  return __builtin_amdgcn_wmma_f32_16x16x32_bf16(false, a, false, b, (short)0, c,
                                                 false, false);
}

// ---------------------------------------------------------------- fp32 -> bf16
__global__ void cvt_f32_bf16(const float* __restrict__ in,
                             bf16_t* __restrict__ out, int n4) {
  int i = blockIdx.x * blockDim.x + threadIdx.x;
  if (i >= n4) return;
  float4 v = ((const float4*)in)[i];
  unsigned long long pk =
        (unsigned long long)pk2bf(v.x, v.y)
      | ((unsigned long long)pk2bf(v.z, v.w) << 32);
  ((unsigned long long*)out)[i] = pk;
}

// ------------------------------------------------- fused Q/K/V projection GEMM
// block = 4 waves; tile 64(M,rows of x) x 64(N,out features); one shared x
// A-fragment drives 3 weight matrices (12 wmma chains per wave per k-step).
__global__ __launch_bounds__(128) void qkv_gemm(
    const bf16_t* __restrict__ xb,
    const bf16_t* __restrict__ wq, const bf16_t* __restrict__ wk,
    const bf16_t* __restrict__ wv,
    bf16_t* __restrict__ Q, bf16_t* __restrict__ K, bf16_t* __restrict__ Vt) {
  int lane = threadIdx.x & 31;
  int wave = threadIdx.x >> 5;
  int mT = blockIdx.x >> 4;          // 64 row tiles
  int nT = blockIdx.x & 15;          // 16 col tiles
  int row0 = mT * 64 + wave * 16;
  int col0 = nT * 64;

  v8f z = {0.f, 0.f, 0.f, 0.f, 0.f, 0.f, 0.f, 0.f};
  v8f aq[4], ak[4], av[4];
#pragma unroll
  for (int j = 0; j < 4; ++j) { aq[j] = z; ak[j] = z; av[j] = z; }

  for (int k0 = 0; k0 < EMB; k0 += 32) {
    v16bf a = load_a_frag(xb + (size_t)row0 * EMB + k0, EMB, lane);
#pragma unroll
    for (int j = 0; j < 4; ++j) {
      size_t wo = (size_t)(col0 + 16 * j) * EMB + k0;
      aq[j] = wmma_bf16(a, load_b_frag(wq + wo, EMB, lane), aq[j]);
      ak[j] = wmma_bf16(a, load_b_frag(wk + wo, EMB, lane), ak[j]);
      av[j] = wmma_bf16(a, load_b_frag(wv + wo, EMB, lane), av[j]);
    }
  }

  int n = lane & 15, hi = lane >> 4;
  int grow0 = row0 + 8 * hi;                 // first token row of this lane half
  int b = grow0 >> 11;                       // whole 8-row group shares batch idx
  int s0w = grow0 & (SEQ - 1);
#pragma unroll
  for (int j = 0; j < 4; ++j) {
    int gcol = col0 + 16 * j + n;            // global feature
    int h = gcol >> 6, d = gcol & (HDIM - 1);
    int bh = b * HEADS + h;
    // Vt: along r the address is contiguous in s -> one 16B packed store
    v4u pk;
#pragma unroll
    for (int r2 = 0; r2 < 4; ++r2)
      pk[r2] = pk2bf(av[j][2 * r2], av[j][2 * r2 + 1]);
    *(v4u*)(Vt + ((size_t)bh * HDIM + d) * SEQ + s0w) = pk;
#pragma unroll
    for (int r = 0; r < 8; ++r) {
      int s = s0w + r;
      Q[((size_t)bh * SEQ + s) * HDIM + d] = f2bf(aq[j][r] * 0.125f); // 1/sqrt(64)
      K[((size_t)bh * SEQ + s) * HDIM + d] = f2bf(ak[j][r]);
    }
  }
}

// --------------------------------------------------------- flash attention
// block = 4 waves; block tile = one (b,h) x 64 q-rows; wave owns 16 q-rows.
// 64-key chunks: 8 score WMMAs + 8 PV WMMAs + 2 row-sum WMMAs (ones trick).
__global__ __launch_bounds__(128) void attn(
    const bf16_t* __restrict__ Q, const bf16_t* __restrict__ K,
    const bf16_t* __restrict__ Vt, bf16_t* __restrict__ ctx) {
  __shared__ __align__(64) bf16_t Plds[4][16 * 64];   // per-wave P tile (2KB each)

  int lane = threadIdx.x & 31;
  int wave = threadIdx.x >> 5;
  int bh = blockIdx.x >> 5;                    // [0,32) = B*H
  int mT = blockIdx.x & 31;                    // 32 q tiles of 64
  int b = bh >> 4, h = bh & 15;
  const bf16_t* Qh = Q + (size_t)bh * SEQ * HDIM;
  const bf16_t* Kh = K + (size_t)bh * SEQ * HDIM;
  const bf16_t* Vh = Vt + (size_t)bh * HDIM * SEQ;
  int m0 = mT * 64 + wave * 16;                // first q row of this wave
  int n = lane & 15, hi = lane >> 4;

  v16bf qa0 = load_a_frag(Qh + (size_t)m0 * HDIM, HDIM, lane);       // d = 0..31
  v16bf qa1 = load_a_frag(Qh + (size_t)m0 * HDIM + 32, HDIM, lane);  // d = 32..63

  FragU of;                                    // all-ones B fragment for row sums
#pragma unroll
  for (int i = 0; i < 8; ++i) of.u8[i] = 0x3F803F80u;   // bf16 1.0 x2
  v16bf ones = of.v;

  v8f z = {0.f, 0.f, 0.f, 0.f, 0.f, 0.f, 0.f, 0.f};
  v8f o0 = z, o1 = z, o2 = z, o3 = z;
  float mst[8], lst[8], lsc[8];
#pragma unroll
  for (int r = 0; r < 8; ++r) { mst[r] = -3.0e38f; lst[r] = 0.f; }

  bf16_t* P = &Plds[wave][0];

  for (int t0 = 0; t0 <= m0 + 15; t0 += 64) {  // causal chunk loop (64 keys)
    v8f s0 = z, s1 = z, s2 = z, s3 = z;
    s0 = wmma_bf16(qa0, load_b_frag(Kh + (size_t)t0 * HDIM,           HDIM, lane), s0);
    s0 = wmma_bf16(qa1, load_b_frag(Kh + (size_t)t0 * HDIM + 32,      HDIM, lane), s0);
    s1 = wmma_bf16(qa0, load_b_frag(Kh + (size_t)(t0+16) * HDIM,      HDIM, lane), s1);
    s1 = wmma_bf16(qa1, load_b_frag(Kh + (size_t)(t0+16) * HDIM + 32, HDIM, lane), s1);
    s2 = wmma_bf16(qa0, load_b_frag(Kh + (size_t)(t0+32) * HDIM,      HDIM, lane), s2);
    s2 = wmma_bf16(qa1, load_b_frag(Kh + (size_t)(t0+32) * HDIM + 32, HDIM, lane), s2);
    s3 = wmma_bf16(qa0, load_b_frag(Kh + (size_t)(t0+48) * HDIM,      HDIM, lane), s3);
    s3 = wmma_bf16(qa1, load_b_frag(Kh + (size_t)(t0+48) * HDIM + 32, HDIM, lane), s3);

#pragma unroll
    for (int r = 0; r < 8; ++r) {
      int row = m0 + r + 8 * hi;
      float v0 = (t0 + n      > row) ? -3.0e38f : s0[r];
      float v1 = (t0 + 16 + n > row) ? -3.0e38f : s1[r];
      float v2 = (t0 + 32 + n > row) ? -3.0e38f : s2[r];
      float v3 = (t0 + 48 + n > row) ? -3.0e38f : s3[r];
      float mx = fmaxf(fmaxf(v0, v1), fmaxf(v2, v3));
#pragma unroll
      for (int off = 8; off > 0; off >>= 1)    // max across 16-lane half
        mx = fmaxf(mx, __shfl_xor(mx, off, 32));
      float mnew = fmaxf(mst[r], mx);
      float sc = __expf(mst[r] - mnew);
      mst[r] = mnew;
      lsc[r] = sc;
      o0[r] *= sc; o1[r] *= sc; o2[r] *= sc; o3[r] *= sc;
      int rl = r + 8 * hi;                     // C-layout -> LDS (A-layout src)
      P[rl * 64 + n]      = f2bf(__expf(v0 - mnew));
      P[rl * 64 + 16 + n] = f2bf(__expf(v1 - mnew));
      P[rl * 64 + 32 + n] = f2bf(__expf(v2 - mnew));
      P[rl * 64 + 48 + n] = f2bf(__expf(v3 - mnew));
    }
    asm volatile("s_wait_dscnt 0" ::: "memory"); // P visible before transpose read
    v16bf pa0 = load_a_frag(P,      64, lane);   // 16x32, k = t0..t0+31
    v16bf pa1 = load_a_frag(P + 32, 64, lane);   // 16x32, k = t0+32..t0+63

    // row sums on the matrix pipe instead of shuffle reductions
    v8f lw = wmma_bf16(pa0, ones, z);
    lw = wmma_bf16(pa1, ones, lw);
#pragma unroll
    for (int r = 0; r < 8; ++r) lst[r] = lst[r] * lsc[r] + lw[r];

    o0 = wmma_bf16(pa0, load_b_frag(Vh + t0,                          SEQ, lane), o0);
    o0 = wmma_bf16(pa1, load_b_frag(Vh + t0 + 32,                     SEQ, lane), o0);
    o1 = wmma_bf16(pa0, load_b_frag(Vh + (size_t)16 * SEQ + t0,       SEQ, lane), o1);
    o1 = wmma_bf16(pa1, load_b_frag(Vh + (size_t)16 * SEQ + t0 + 32,  SEQ, lane), o1);
    o2 = wmma_bf16(pa0, load_b_frag(Vh + (size_t)32 * SEQ + t0,       SEQ, lane), o2);
    o2 = wmma_bf16(pa1, load_b_frag(Vh + (size_t)32 * SEQ + t0 + 32,  SEQ, lane), o2);
    o3 = wmma_bf16(pa0, load_b_frag(Vh + (size_t)48 * SEQ + t0,       SEQ, lane), o3);
    o3 = wmma_bf16(pa1, load_b_frag(Vh + (size_t)48 * SEQ + t0 + 32,  SEQ, lane), o3);
  }

#pragma unroll
  for (int r = 0; r < 8; ++r) {
    float inv = 1.0f / lst[r];
    int s = m0 + r + 8 * hi;
    size_t base = ((size_t)b * SEQ + s) * EMB + h * HDIM + n;
    ctx[base + 0]  = f2bf(o0[r] * inv);
    ctx[base + 16] = f2bf(o1[r] * inv);
    ctx[base + 32] = f2bf(o2[r] * inv);
    ctx[base + 48] = f2bf(o3[r] * inv);
  }
}

// ------------------------------------------------------- output projection
__global__ __launch_bounds__(128) void out_gemm(
    const bf16_t* __restrict__ cb, const bf16_t* __restrict__ wu,
    const float* __restrict__ bu, float* __restrict__ out) {
  int lane = threadIdx.x & 31;
  int wave = threadIdx.x >> 5;
  int mT = blockIdx.x >> 4;
  int nT = blockIdx.x & 15;
  int row0 = mT * 64 + wave * 16;
  int col0 = nT * 64;

  v8f z = {0.f, 0.f, 0.f, 0.f, 0.f, 0.f, 0.f, 0.f};
  v8f acc[4];
#pragma unroll
  for (int j = 0; j < 4; ++j) acc[j] = z;

  for (int k0 = 0; k0 < EMB; k0 += 32) {
    v16bf a = load_a_frag(cb + (size_t)row0 * EMB + k0, EMB, lane);
#pragma unroll
    for (int j = 0; j < 4; ++j) {
      size_t wo = (size_t)(col0 + 16 * j) * EMB + k0;
      acc[j] = wmma_bf16(a, load_b_frag(wu + wo, EMB, lane), acc[j]);
    }
  }

  int n = lane & 15, hi = lane >> 4;
#pragma unroll
  for (int j = 0; j < 4; ++j) {
#pragma unroll
    for (int r = 0; r < 8; ++r) {
      int grow = row0 + r + 8 * hi;
      int gcol = col0 + 16 * j + n;
      out[(size_t)grow * EMB + gcol] = acc[j][r] + bu[gcol];
    }
  }
}

// ------------------------------------------------------------------ launch
extern "C" void kernel_launch(void* const* d_in, const int* in_sizes, int n_in,
                              void* d_out, int out_size, void* d_ws, size_t ws_size,
                              hipStream_t stream) {
  const float* x  = (const float*)d_in[0];
  const float* Wk = (const float*)d_in[1];
  const float* Wq = (const float*)d_in[2];
  const float* Wv = (const float*)d_in[3];
  const float* Wu = (const float*)d_in[4];
  const float* bu = (const float*)d_in[5];

  char* ws = (char*)d_ws;
  const size_t MB = 1024 * 1024;
  bf16_t* xb  = (bf16_t*)(ws +  0 * MB);  // 8 MB  x in bf16
  bf16_t* wkb = (bf16_t*)(ws +  8 * MB);  // 2 MB
  bf16_t* wqb = (bf16_t*)(ws + 10 * MB);  // 2 MB
  bf16_t* wvb = (bf16_t*)(ws + 12 * MB);  // 2 MB
  bf16_t* wub = (bf16_t*)(ws + 14 * MB);  // 2 MB
  bf16_t* Qb  = (bf16_t*)(ws + 16 * MB);  // 8 MB  [B,H,S,D] (pre-scaled)
  bf16_t* Kb  = (bf16_t*)(ws + 24 * MB);  // 8 MB  [B,H,S,D]
  bf16_t* Vtb = (bf16_t*)(ws + 32 * MB);  // 8 MB  [B,H,D,S]
  bf16_t* cxb = (bf16_t*)(ws + 40 * MB);  // 8 MB  [B,S,E]   (48 MB total)

  const int nx = NROW * EMB;   // 4,194,304
  const int nw = EMB * EMB;    // 1,048,576
  cvt_f32_bf16<<<(nx / 4 + 255) / 256, 256, 0, stream>>>(x,  xb,  nx / 4);
  cvt_f32_bf16<<<(nw / 4 + 255) / 256, 256, 0, stream>>>(Wk, wkb, nw / 4);
  cvt_f32_bf16<<<(nw / 4 + 255) / 256, 256, 0, stream>>>(Wq, wqb, nw / 4);
  cvt_f32_bf16<<<(nw / 4 + 255) / 256, 256, 0, stream>>>(Wv, wvb, nw / 4);
  cvt_f32_bf16<<<(nw / 4 + 255) / 256, 256, 0, stream>>>(Wu, wub, nw / 4);

  qkv_gemm<<<1024, 128, 0, stream>>>(xb, wqb, wkb, wvb, Qb, Kb, Vtb);
  attn<<<1024, 128, 0, stream>>>(Qb, Kb, Vtb, cxb);
  out_gemm<<<1024, 128, 0, stream>>>(cxb, wub, bu, (float*)d_out);
}
